// Encoder_46033459478624
// MI455X (gfx1250) — matrix operand order repaired and checked by reading, not compiled, not run on hardware
//
#include <hip/hip_runtime.h>
#include <cmath>

#define B_  8
#define T_  16
#define CIN 3
#define H_  32
#define W_  32
#define HID 64
#define KS  7
#define PAD 3
#define HP  38
#define WP  38

typedef _Float16 v16h __attribute__((ext_vector_type(16)));
typedef _Float16 v8h  __attribute__((ext_vector_type(8)));
typedef _Float16 v4h  __attribute__((ext_vector_type(4)));
typedef float    v8f  __attribute__((ext_vector_type(8)));

union V16 { v16h v; v8h h8[2]; v4h h4[4]; };

__device__ __forceinline__ v8f wmma_f16(v16h a, v16h b, v8f c) {
  // 8 args: (neg_a, A, neg_b, B, c_mod, C, reuse_a, reuse_b)
  return __builtin_amdgcn_wmma_f32_16x16x32_f16(false, a, false, b, (short)0, c,
                                                false, false);
}

__device__ __forceinline__ float sigmoidf_(float x) {
  return 1.0f / (1.0f + __expf(-x));
}

// ---------------------------------------------------------------------------
// Pack x (B,T,C,H,W) f32 -> spatially padded, channel-last (B,T,38,38,4) f16.
// Channel 3 is a zero pad so (kw,c) flattens to a clean K index per kh.
// ---------------------------------------------------------------------------
__global__ __launch_bounds__(256) void pack_x(const float* __restrict__ x,
                                              _Float16* __restrict__ xp) {
  const int idx = blockIdx.x * 256 + threadIdx.x;      // one padded pixel
  const int total = B_ * T_ * HP * WP;
  if (idx >= total) return;
  int r = idx;
  const int wp = r % WP; r /= WP;
  const int hp = r % HP; r /= HP;
  const int t  = r % T_;
  const int b  = r / T_;
  v4h v = {};
  const int hh = hp - PAD, ww = wp - PAD;
  if (hh >= 0 && hh < H_ && ww >= 0 && ww < W_) {
    const float* px = x + (((size_t)(b * T_ + t) * CIN) * H_ + hh) * W_ + ww;
    v[0] = (_Float16)px[0 * H_ * W_];
    v[1] = (_Float16)px[1 * H_ * W_];
    v[2] = (_Float16)px[2 * H_ * W_];
  }
  *(v4h*)(xp + (size_t)idx * 4) = v;
}

// ---------------------------------------------------------------------------
// Pack w_ih into per-lane WMMA B-fragment order.
// Fragment flatten: (((d*7+kh)*16 + ntile)*512) + lane*16 + j
// K index kk = (lane>>4)*8 + (j&7) + (j>>3)*16 ; kk -> (kw = kk>>2, c = kk&3); kk>=28 or c==3 -> 0
// Value = w[gate = ntile*16 + (lane&15)][c][kh][kw]
// ---------------------------------------------------------------------------
__global__ __launch_bounds__(256) void pack_wih(const float* __restrict__ wf,
                                                const float* __restrict__ wb,
                                                _Float16* __restrict__ wp) {
  const int idx = blockIdx.x * 256 + threadIdx.x;      // total = 2*7*16*512
  const int e    = idx & 511;
  const int lane = e >> 4;
  const int j    = e & 15;
  int r = idx >> 9;
  const int ntile = r & 15; r >>= 4;
  const int kh = r % 7;
  const int d  = r / 7;
  const int kk = ((lane >> 4) << 3) + (j & 7) + ((j >> 3) << 4);
  const int kw = kk >> 2;
  const int c  = kk & 3;
  const int gate = (ntile << 4) + (lane & 15);
  const float* src = d ? wb : wf;
  float v = 0.0f;
  if (kk < 28 && c < CIN)
    v = src[(((size_t)gate * CIN + c) * KS + kh) * KS + kw];
  wp[idx] = (_Float16)v;
}

// ---------------------------------------------------------------------------
// Pack w_hh: flatten ((((d*7+kh)*7+kw)*2+kb)*16 + ntile)*512 + lane*16 + j
// cin = kb*32 + (lane>>4)*8 + (j&7) + (j>>3)*16 ; gate = ntile*16 + (lane&15)
// ---------------------------------------------------------------------------
__global__ __launch_bounds__(256) void pack_whh(const float* __restrict__ wf,
                                                const float* __restrict__ wb,
                                                _Float16* __restrict__ wp) {
  const int idx = blockIdx.x * 256 + threadIdx.x;      // total = 2*7*7*2*16*512
  const int e    = idx & 511;
  const int lane = e >> 4;
  const int j    = e & 15;
  int r = idx >> 9;
  const int ntile = r & 15; r >>= 4;
  const int kb = r & 1;    r >>= 1;
  const int kw = r % 7;    r /= 7;
  const int kh = r % 7;
  const int d  = r / 7;
  const int cin  = (kb << 5) + ((lane >> 4) << 3) + (j & 7) + ((j >> 3) << 4);
  const int gate = (ntile << 4) + (lane & 15);
  const float* src = d ? wb : wf;
  wp[idx] = (_Float16)src[((size_t)gate * HID + cin) * (KS * KS) + kh * KS + kw];
}

// ---------------------------------------------------------------------------
// Zero h-state (padded, f16) and c-state (f32). Word counts passed in.
// ---------------------------------------------------------------------------
__global__ __launch_bounds__(256) void zero_state(unsigned* hp, int hpw,
                                                  unsigned* cs, int csw) {
  const int i = blockIdx.x * 256 + threadIdx.x;
  if (i < hpw) hp[i] = 0u;
  if (i < csw) cs[i] = 0u;
}

// ---------------------------------------------------------------------------
// One recurrent timestep, both directions.
// grid = 1024 blocks (d,b,h,w0-tile), 128 threads = 4 waves.
// Wave w computes gates [64w, 64w+64) for 16 pixels via 105 K=32 WMMA blocks.
// The 49 hidden shifts are software-pipelined 2-deep: the next shift's 12
// fragment loads are issued BEFORE the current shift's 8 WMMAs, with a
// sched_barrier pinning that order so the backend cannot re-serialize.
// amdgpu_waves_per_eu(2) relaxes the VGPR budget for the double buffers.
// ---------------------------------------------------------------------------
__global__ __launch_bounds__(128)
__attribute__((amdgpu_waves_per_eu(2)))
void lstm_step(
    const _Float16* __restrict__ xp,
    const _Float16* __restrict__ wihp,
    const _Float16* __restrict__ whhp,
    const float* __restrict__ bf,
    const float* __restrict__ bb,
    _Float16* __restrict__ hpad,
    float* __restrict__ cst,
    float* __restrict__ out,
    int s)
{
  __shared__ float gbuf[16 * 256];

  const int blk = blockIdx.x;
  const int d   = blk >> 9;
  const int b   = (blk >> 6) & 7;
  const int h   = (blk >> 1) & 31;
  const int w0  = (blk & 1) << 4;
  const int t   = d ? (T_ - 1 - s) : s;

  const int lane = threadIdx.x & 31;
  const int wave = threadIdx.x >> 5;
  const int m    = lane & 15;        // A row (pixel) / C-D column (gate)
  const int hs   = lane >> 4;        // lane-half select
  const int h8   = hs << 3;
  const int ntb  = wave << 2;        // first n-tile of this wave
  const int nb   = wave << 6;        // first gate of this wave

  // Accumulators initialized with bias (same value in all 8 rows of a column).
  const float* pbias = d ? bb : bf;
  v8f acc0, acc1, acc2, acc3;
  {
    const float b0 = pbias[nb + m], b1 = pbias[nb + 16 + m];
    const float b2 = pbias[nb + 32 + m], b3 = pbias[nb + 48 + m];
#pragma unroll
    for (int r = 0; r < 8; ++r) { acc0[r] = b0; acc1[r] = b1; acc2[r] = b2; acc3[r] = b3; }
  }

  const _Float16* xbase = xp   + (size_t)(b * T_ + t) * (HP * WP * 4);
  const _Float16* hbase = hpad + (size_t)(d * B_ + b) * (HP * WP * HID);

  // Double-buffered hidden-shift fragments (2 A frags + 8 B frags per shift).
  V16  pa0[2], pa1[2];
  v16h pbf[2][8];

  auto load_shift = [&](int kh_, int kw_, int buf) {
    const _Float16* hpix = hbase + (size_t)(h + kh_) * (WP * HID) +
                           (size_t)(w0 + m + kw_) * HID;
    pa0[buf].h8[0] = *(const v8h*)(hpix + h8);
    pa0[buf].h8[1] = *(const v8h*)(hpix + 16 + h8);
    pa1[buf].h8[0] = *(const v8h*)(hpix + 32 + h8);
    pa1[buf].h8[1] = *(const v8h*)(hpix + 48 + h8);
    const _Float16* wq = whhp +
        ((((size_t)((d * KS + kh_) * KS + kw_) * 2) * 16) + ntb) * 512 +
        (size_t)lane * 16;
#pragma unroll
    for (int i = 0; i < 4; ++i) pbf[buf][i]     = *(const v16h*)(wq + i * 512);
#pragma unroll
    for (int i = 0; i < 4; ++i) pbf[buf][4 + i] = *(const v16h*)(wq + 8192 + i * 512);
  };

  for (int kh = 0; kh < KS; ++kh) {
    // Prologue: issue loads for hidden shift (kh, kw=0) first; their latency
    // overlaps the input-conv block below.
    load_shift(kh, 0, 0);
    __builtin_amdgcn_sched_barrier(0);

    // ---- input->gates: one K=32 block (kw 0..6 x c 0..3, 28 valid) ----
    {
      const _Float16* prow = xbase + (size_t)(h + kh) * (WP * 4) + (size_t)(w0 + m) * 4;
      V16 a;
      a.h4[0] = *(const v4h*)(prow + h8);
      a.h4[1] = *(const v4h*)(prow + h8 + 4);
      a.h4[2] = *(const v4h*)(prow + 16 + h8);
      const v4h tail = *(const v4h*)(prow + 16 + h8 + 4);  // kk 28..31 for hs==1
      const v4h z = {};
      a.h4[3] = hs ? z : tail;   // discard OOB K (select, NOT multiply: NaN-safe)
      const _Float16* wb = wihp + ((size_t)(d * KS + kh) * 16 + ntb) * 512 + lane * 16;
      const v16h b0 = *(const v16h*)(wb);
      const v16h b1 = *(const v16h*)(wb + 512);
      const v16h b2 = *(const v16h*)(wb + 1024);
      const v16h b3 = *(const v16h*)(wb + 1536);
      acc0 = wmma_f16(a.v, b0, acc0);
      acc1 = wmma_f16(a.v, b1, acc1);
      acc2 = wmma_f16(a.v, b2, acc2);
      acc3 = wmma_f16(a.v, b3, acc3);
    }

    // ---- hidden->gates: 7 shifts, 2-deep pipelined ----
#pragma unroll
    for (int kw = 0; kw < KS; ++kw) {
      const int cb = kw & 1;
      if (kw < KS - 1) load_shift(kh, kw + 1, cb ^ 1);
      // Pin order: next-shift loads above stay ahead of this shift's WMMAs.
      __builtin_amdgcn_sched_barrier(0);
      acc0 = wmma_f16(pa0[cb].v, pbf[cb][0], acc0);
      acc1 = wmma_f16(pa0[cb].v, pbf[cb][1], acc1);
      acc2 = wmma_f16(pa0[cb].v, pbf[cb][2], acc2);
      acc3 = wmma_f16(pa0[cb].v, pbf[cb][3], acc3);
      acc0 = wmma_f16(pa1[cb].v, pbf[cb][4], acc0);
      acc1 = wmma_f16(pa1[cb].v, pbf[cb][5], acc1);
      acc2 = wmma_f16(pa1[cb].v, pbf[cb][6], acc2);
      acc3 = wmma_f16(pa1[cb].v, pbf[cb][7], acc3);
    }
  }

  // Spill gate pre-activations: C/D layout -> lane holds col N=m, row = h8+r.
#pragma unroll
  for (int r = 0; r < 8; ++r) {
    const int row = h8 + r;
    gbuf[row * 256 + nb + m]      = acc0[r];
    gbuf[row * 256 + nb + 16 + m] = acc1[r];
    gbuf[row * 256 + nb + 32 + m] = acc2[r];
    gbuf[row * 256 + nb + 48 + m] = acc3[r];
  }
  __syncthreads();

  // Fused LSTM update: 16 pixels x 64 channels / 128 threads = 8 each.
#pragma unroll
  for (int k = 0; k < 8; ++k) {
    const int idx = threadIdx.x + (k << 7);
    const int mm  = idx >> 6;
    const int ch  = idx & 63;
    const float gi = gbuf[mm * 256 + ch];
    const float gf = gbuf[mm * 256 + 64 + ch];
    const float gg = gbuf[mm * 256 + 128 + ch];
    const float go = gbuf[mm * 256 + 192 + ch];
    const size_t cidx = (((size_t)(d * B_ + b) * H_ + h) * W_ + (w0 + mm)) * HID + ch;
    const float cp = cst[cidx];
    const float cn = sigmoidf_(gf) * cp + sigmoidf_(gi) * tanhf(gg);
    const float hn = sigmoidf_(go) * tanhf(cn);
    cst[cidx] = cn;
    hpad[((size_t)(d * B_ + b) * HP + (h + PAD)) * (size_t)(WP * HID) +
         (size_t)(w0 + mm + PAD) * HID + ch] = (_Float16)hn;
    out[(((size_t)(b * T_ + t) * (2 * HID) + d * HID + ch) * H_ + h) * W_ + (w0 + mm)] = hn;
  }
}

// ---------------------------------------------------------------------------
extern "C" void kernel_launch(void* const* d_in, const int* in_sizes, int n_in,
                              void* d_out, int out_size, void* d_ws, size_t ws_size,
                              hipStream_t stream) {
  (void)in_sizes; (void)n_in; (void)out_size; (void)ws_size;
  const float* x     = (const float*)d_in[0];
  const float* wih_f = (const float*)d_in[1];
  const float* whh_f = (const float*)d_in[2];
  const float* b_f   = (const float*)d_in[3];
  const float* wih_b = (const float*)d_in[4];
  const float* whh_b = (const float*)d_in[5];
  const float* b_b   = (const float*)d_in[6];
  float* out = (float*)d_out;

  // Workspace carve-up (256B aligned slabs, +guard for 8B A-fragment overreads)
  const size_t XP_HALFS   = (size_t)B_ * T_ * HP * WP * 4;         //   739,328
  const size_t HPAD_HALFS = (size_t)2 * B_ * HP * WP * HID;        // 1,478,656
  const size_t CST_FLOATS = (size_t)2 * B_ * H_ * W_ * HID;        // 1,048,576
  const size_t WIH_HALFS  = (size_t)2 * KS * 16 * 512;             //   114,688
  const size_t WHH_HALFS  = (size_t)2 * KS * KS * 2 * 16 * 512;    // 1,605,632

  char* ws = (char*)d_ws;
  size_t off = 0;
  auto carve = [&](size_t bytes) -> void* {
    void* p = ws + off;
    off += (bytes + 255) & ~(size_t)255;
    return p;
  };
  _Float16* xp   = (_Float16*)carve(XP_HALFS * 2 + 64);
  _Float16* hpad = (_Float16*)carve(HPAD_HALFS * 2 + 64);
  float*    cst  = (float*)   carve(CST_FLOATS * 4);
  _Float16* wihp = (_Float16*)carve(WIH_HALFS * 2);
  _Float16* whhp = (_Float16*)carve(WHH_HALFS * 2);

  pack_x  <<<(B_ * T_ * HP * WP + 255) / 256, 256, 0, stream>>>(x, xp);
  pack_wih<<<(int)(WIH_HALFS / 256), 256, 0, stream>>>(wih_f, wih_b, wihp);
  pack_whh<<<(int)(WHH_HALFS / 256), 256, 0, stream>>>(whh_f, whh_b, whhp);
  {
    const int hpw = (int)(HPAD_HALFS / 2);   // u32 words of f16 state
    const int csw = (int)CST_FLOATS;
    const int mx  = hpw > csw ? hpw : csw;
    zero_state<<<(mx + 255) / 256, 256, 0, stream>>>((unsigned*)hpad, hpw,
                                                     (unsigned*)cst, csw);
  }
  for (int s = 0; s < T_; ++s)
    lstm_step<<<1024, 128, 0, stream>>>(xp, wihp, whhp, b_f, b_b,
                                        hpad, cst, out, s);
}
